// SAGEConv_343597384517
// MI455X (gfx1250) — matrix operand order repaired and checked by reading, not compiled, not run on hardware
//
#include <hip/hip_runtime.h>

typedef __attribute__((ext_vector_type(16))) _Float16 v16h;
typedef __attribute__((ext_vector_type(8)))  float    v8f;

#define SA_STRIDE 136  // 128 cols + 8 pad (even alignment, no bank conflicts)
#define SW_STRIDE 136

// ---------------------------------------------------------------------------
// Edge gather + scatter-add (the bandwidth-dominant phase).
// thread = (edge, 4-feat group): float4 gather (256B/edge coalesced),
// 4x global_atomic_add_f32 scatter; all tables L2-resident (192MB L2).
// Degree count folded into the first pass (lane f==0 per edge).
// ---------------------------------------------------------------------------
__global__ void sage_agg_kernel(const float* __restrict__ feat,
                                const int* __restrict__ src,
                                const int* __restrict__ dst,
                                float* __restrict__ agg,
                                float* __restrict__ deg,
                                int n_edges, int add_deg) {
  int i = blockIdx.x * blockDim.x + threadIdx.x;
  int total = n_edges << 4;
  if (i >= total) return;
  int e = i >> 4;
  int f = (i & 15) << 2;
  int s = src[e];
  int d = dst[e];
  const float4 v = *reinterpret_cast<const float4*>(feat + (size_t)s * 64 + f);
  float* o = agg + (size_t)d * 64 + f;
  atomicAdd(o + 0, v.x);
  atomicAdd(o + 1, v.y);
  atomicAdd(o + 2, v.z);
  atomicAdd(o + 3, v.w);
  if (add_deg && (i & 15) == 0) atomicAdd(&deg[d], 1.0f);
}

// ---------------------------------------------------------------------------
// Fused SAGE dense layer: out = act( [feat | agg/deg] @ [Wself; Wneigh] + b )
// Block = 128 threads = 4 waves; each wave computes a 16-row stripe with
// v_wmma_f32_16x16x32_f16 (4 K-steps of 32, N_OUT/16 column tiles, fully
// unrolled at compile time).
// ---------------------------------------------------------------------------
template <bool RELU, int N_OUT>
__global__ __launch_bounds__(128) void sage_layer_kernel(
    const float* __restrict__ feat, const float* __restrict__ agg,
    const float* __restrict__ deg, const float* __restrict__ Wself,
    const float* __restrict__ Wneigh, const float* __restrict__ bias,
    float* __restrict__ out, int n_nodes) {
  __shared__ _Float16 sA[64 * SA_STRIDE];
  __shared__ _Float16 sW[64 * SW_STRIDE];

  constexpr int NTILES = N_OUT / 16;

  const int tid = threadIdx.x;
  const int rb = blockIdx.x * 64;
  const bool fullTile = (rb + 64 <= n_nodes);

  // Stage weights transposed: sW[n][k], k in [0,128): rows 0..63 = Wself,
  // 64..127 = Wneigh. float4 along n (contiguous in W), f16 scatter in LDS.
  {
    constexpr int N4CNT = N_OUT / 4;  // float4 groups per W row
#pragma unroll 1
    for (int idx = tid; idx < 32 * N_OUT; idx += 128) {  // (128*N_OUT)/4 elems
      int k = idx / N4CNT;
      int n4 = (idx - k * N4CNT) << 2;
      const float* Wp = (k < 64) ? (Wself + (size_t)k * N_OUT + n4)
                                 : (Wneigh + (size_t)(k - 64) * N_OUT + n4);
      float4 w = *reinterpret_cast<const float4*>(Wp);
      sW[(n4 + 0) * SW_STRIDE + k] = (_Float16)w.x;
      sW[(n4 + 1) * SW_STRIDE + k] = (_Float16)w.y;
      sW[(n4 + 2) * SW_STRIDE + k] = (_Float16)w.z;
      sW[(n4 + 3) * SW_STRIDE + k] = (_Float16)w.w;
    }
  }

  // Stage A rows (float4): cols 0..63 = feat[node], 64..127 = agg[node]/deg.
#pragma unroll 1
  for (int idx = tid; idx < 64 * 32; idx += 128) {
    int i = idx >> 5;
    int c4 = (idx & 31) << 2;
    int node = rb + i;
    float4 v = make_float4(0.f, 0.f, 0.f, 0.f);
    if (node < n_nodes) {
      if (c4 < 64) {
        v = *reinterpret_cast<const float4*>(feat + (size_t)node * 64 + c4);
      } else {
        float inv = 1.0f / fmaxf(deg[node], 1.0f);
        float4 a = *reinterpret_cast<const float4*>(agg + (size_t)node * 64 + (c4 - 64));
        v = make_float4(a.x * inv, a.y * inv, a.z * inv, a.w * inv);
      }
    }
    _Float16* dp = &sA[i * SA_STRIDE + c4];
    dp[0] = (_Float16)v.x;
    dp[1] = (_Float16)v.y;
    dp[2] = (_Float16)v.z;
    dp[3] = (_Float16)v.w;
  }
  __syncthreads();

  const int wave   = tid >> 5;
  const int lane   = tid & 31;
  const int laneM  = lane & 15;
  const int laneHi = lane >> 4;  // 0 or 1
  const int rowOff = wave * 16;

  // Preload A fragments (K = 128 in 4 steps of 32), CDNA5 16-bit A 16x32
  // layout: lanes 0-15 hold K {0..7, 16..23}; lanes 16-31 the same +8.
  v16h afr[4];
#pragma unroll
  for (int kk = 0; kk < 4; ++kk) {
    const _Float16* p = &sA[(rowOff + laneM) * SA_STRIDE + kk * 32 + laneHi * 8];
    v16h a;
#pragma unroll
    for (int j = 0; j < 4; ++j) {
      a[2 * j]         = p[2 * j];
      a[2 * j + 1]     = p[2 * j + 1];
      a[8 + 2 * j]     = p[16 + 2 * j];
      a[8 + 2 * j + 1] = p[16 + 2 * j + 1];
    }
    afr[kk] = a;
  }

  // Preload bias (keeps global loads/waits out of the WMMA loop).
  float bv[NTILES];
#pragma unroll
  for (int nt = 0; nt < NTILES; ++nt) bv[nt] = bias[nt * 16 + laneM];

#pragma unroll
  for (int nt = 0; nt < NTILES; ++nt) {
    v8f acc = {};
#pragma unroll
    for (int kk = 0; kk < 4; ++kk) {
      // B 32x16 layout: col = lane%16; lanes 0-15 hold K {0..7, 16..23}; lanes 16-31 the same +8.
      const _Float16* q = &sW[(nt * 16 + laneM) * SW_STRIDE + kk * 32 + laneHi * 8];
      v16h b;
#pragma unroll
      for (int j = 0; j < 8; ++j) {
        b[j]     = q[j];
        b[8 + j] = q[16 + j];
      }
      acc = __builtin_amdgcn_wmma_f32_16x16x32_f16(
          false, afr[kk], false, b, (short)0, acc, false, false);
    }
    // C/D layout: VGPR r -> row r + 8*laneHi, col = laneM.
    const int colBase = nt * 16 + laneM;
    const int rowBase = rb + rowOff + laneHi * 8;
    if (fullTile) {
#pragma unroll
      for (int r = 0; r < 8; ++r) {
        float v = acc[r] + bv[nt];
        if (RELU) v = fmaxf(v, 0.0f);
        out[(size_t)(rowBase + r) * N_OUT + colBase] = v;
      }
    } else {
#pragma unroll
      for (int r = 0; r < 8; ++r) {
        if (rowBase + r < n_nodes) {
          float v = acc[r] + bv[nt];
          if (RELU) v = fmaxf(v, 0.0f);
          out[(size_t)(rowBase + r) * N_OUT + colBase] = v;
        }
      }
    }
  }
}

// ---------------------------------------------------------------------------
// memsets -> agg(x)+deg -> layer1(WMMA,relu) -> agg(h1) -> layer2(WMMA)
// ---------------------------------------------------------------------------
extern "C" void kernel_launch(void* const* d_in, const int* in_sizes, int n_in,
                              void* d_out, int out_size, void* d_ws, size_t ws_size,
                              hipStream_t stream) {
  const float* x   = (const float*)d_in[0];
  const int*   src = (const int*)d_in[1];
  const int*   dst = (const int*)d_in[2];
  const float* Ws1 = (const float*)d_in[3];
  const float* Wn1 = (const float*)d_in[4];
  const float* b1  = (const float*)d_in[5];
  const float* Ws2 = (const float*)d_in[6];
  const float* Wn2 = (const float*)d_in[7];
  const float* b2  = (const float*)d_in[8];
  float* out = (float*)d_out;

  const int n_nodes = in_sizes[0] / 64;  // 50000
  const int n_edges = in_sizes[1];       // 800000

  float* agg1 = (float*)d_ws;
  float* agg2 = agg1 + (size_t)n_nodes * 64;
  float* h1   = agg2 + (size_t)n_nodes * 64;
  float* deg  = h1 + (size_t)n_nodes * 64;

  hipMemsetAsync(agg1, 0, sizeof(float) * (size_t)n_nodes * 64, stream);
  hipMemsetAsync(agg2, 0, sizeof(float) * (size_t)n_nodes * 64, stream);
  hipMemsetAsync(deg, 0, sizeof(float) * (size_t)n_nodes, stream);

  const int tpb = 256;
  const int agg_threads = n_edges * 16;
  const int agg_blocks = (agg_threads + tpb - 1) / tpb;
  const int layer_blocks = (n_nodes + 63) / 64;

  sage_agg_kernel<<<agg_blocks, tpb, 0, stream>>>(x, src, dst, agg1, deg, n_edges, 1);

  sage_layer_kernel<true, 64><<<layer_blocks, 128, 0, stream>>>(x, agg1, deg, Ws1, Wn1,
                                                                b1, h1, n_nodes);

  sage_agg_kernel<<<agg_blocks, tpb, 0, stream>>>(h1, src, dst, agg2, deg, n_edges, 0);

  sage_layer_kernel<false, 16><<<layer_blocks, 128, 0, stream>>>(h1, agg2, deg, Ws2,
                                                                 Wn2, b2, out, n_nodes);
}
